// GATNet_8083128451557
// MI455X (gfx1250) — matrix operand order repaired and checked by reading, not compiled, not run on hardware
//
#include <hip/hip_runtime.h>
#include <hip/hip_bf16.h>

// ---------------- problem constants (fixed by the reference) ----------------
constexpr int N_  = 20000;
constexpr int G_  = 64;
constexpr int H_  = 4;
constexpr int C_  = 64;
constexpr int IN_ = 128;
constexpr int E_  = 640000;
constexpr int P_  = 10;
constexpr int D_  = H_ * C_;        // 256
constexpr int EH_ = E_ + N_;        // edges incl. self loops = 660000

// ---------------- output layout (floats, concat in return order) -----------
constexpr int OFF_LOGITS = 0;                       // (64,2)
constexpr int OFF_PROBS  = OFF_LOGITS + G_ * 2;     // (64,2)
constexpr int OFF_ACTIVE = OFF_PROBS + G_ * 2;      // (20000,) as 0/1 floats
constexpr int OFF_GE     = OFF_ACTIVE + N_;         // (64,256)
constexpr int OFF_KL     = OFF_GE + G_ * D_;        // scalar
constexpr int OFF_POS    = OFF_KL + 1;              // scalar
constexpr int OFF_SIM    = OFF_POS + 1;             // (64,10)
constexpr int OFF_DIST   = OFF_SIM + G_ * P_;       // (64,10)

// ---------------- workspace layout (float offsets) --------------------------
constexpr size_t WS_HPRE  = 0;                        // N*D (pre-attn h; reused as fc1 output t)
constexpr size_t WS_H1    = WS_HPRE  + (size_t)N_*D_; // N*D
constexpr size_t WS_NF    = WS_H1    + (size_t)N_*D_; // N*D
constexpr size_t WS_SSRC  = WS_NF    + (size_t)N_*D_; // N*H
constexpr size_t WS_SDST  = WS_SSRC  + (size_t)N_*H_;
constexpr size_t WS_NMAX  = WS_SDST  + (size_t)N_*H_;
constexpr size_t WS_NSUM  = WS_NMAX  + (size_t)N_*H_;
constexpr size_t WS_EBUF  = WS_NSUM  + (size_t)N_*H_; // EH*H
constexpr size_t WS_ASSN  = WS_EBUF  + (size_t)EH_*H_; // N*2
constexpr size_t WS_GUM   = WS_ASSN  + (size_t)N_*2;   // N*2
constexpr size_t WS_FMEAN = WS_GUM   + (size_t)N_*2;   // D
constexpr size_t WS_FSTD  = WS_FMEAN + D_;             // D
constexpr size_t WS_GSUM  = WS_FSTD  + D_;             // G*D
constexpr size_t WS_CNT   = WS_GSUM  + (size_t)G_*D_;  // G
constexpr size_t WS_AA    = WS_CNT   + G_;             // N*2
constexpr size_t WS_SCAL  = WS_AA    + (size_t)N_*2;   // 16 scalars

// ---------------- vector types -----------------------------------------------
typedef __attribute__((ext_vector_type(16))) _Float16 v16h;
typedef __attribute__((ext_vector_type(8)))  float    v8f;
typedef __attribute__((ext_vector_type(4)))  unsigned u32x4;
typedef __attribute__((ext_vector_type(4)))  int      i32x4;
typedef __attribute__((ext_vector_type(8)))  int      i32x8;

#if defined(__has_builtin)
#if __has_builtin(__builtin_amdgcn_tensor_load_to_lds) && __has_builtin(__builtin_amdgcn_s_wait_tensorcnt)
#define HAVE_TDM 1
#endif
#endif
#ifndef HAVE_TDM
#define HAVE_TDM 0
#endif

// ---------------- device helpers --------------------------------------------
__device__ __forceinline__ float atomicMaxFloat(float* addr, float val) {
  unsigned int* a = (unsigned int*)addr;
  unsigned int old = *a;
  for (;;) {
    float f = __uint_as_float(old);
    if (f >= val) break;
    unsigned int assumed = old;
    old = atomicCAS(a, assumed, __float_as_uint(val));
    if (old == assumed) break;
  }
  return __uint_as_float(old);
}

__device__ __forceinline__ float urand(unsigned a, unsigned b, unsigned seed) {
  unsigned s = a * 747796405u + b * 2891336453u + seed;
  s ^= s >> 17; s *= 0xED5AD4BBu;
  s ^= s >> 11; s *= 0xAC4C1B51u;
  s ^= s >> 15; s *= 0x31848BABu;
  s ^= s >> 14;
  return (float)(s >> 8) * (1.0f / 16777216.0f);
}

__device__ __forceinline__ void edge_sd(const int* __restrict__ ei, int idx, int& s, int& d) {
  if (idx < E_) { s = ei[idx]; d = ei[E_ + idx]; }
  else          { s = idx - E_; d = idx - E_; }
}

// ---------------- generic fill ----------------------------------------------
__global__ void k_fill(float* __restrict__ p, float v, int n) {
  int i = blockIdx.x * blockDim.x + threadIdx.x;
  if (i < n) p[i] = v;
}

// =============================================================================
// WMMA GEMM: C[N_, D_] = act(A[N_, K] @ W[D_, K]^T + bias)
//  * grid = (ceil((N_/32)/8), D_/16); block = 256 threads (8 waves)
//  * each block owns one 16-column strip; the 16 x K weight tile is staged
//    into LDS once per block via the Tensor Data Mover (tensor_load_to_lds,
//    2D descriptor, TENSORcnt-waited), falling back to a cooperative copy.
//  * each wave computes a 32x16 output tile: 2 x v_wmma_f32_16x16x32_f16 per
//    K-step, K-loop fully unrolled (K is a template parameter).
// =============================================================================
template <int K, int ACT>
__global__ void k_wmma_gemm(const float* __restrict__ A,
                            const float* __restrict__ W,
                            const float* __restrict__ bias,
                            float* __restrict__ C) {
  __shared__ float smem[16 * K];               // weight tile: row r (=col) * K + k
  const int tn   = blockIdx.y;                 // column-tile index
  const int lane = threadIdx.x & 31;
  const int tmBig = blockIdx.x * 8 + (threadIdx.x >> 5);  // 32-row slab index

  // ---- stage W[tn*16 .. tn*16+15][0..K) into LDS ----
#if HAVE_TDM
  if (threadIdx.x < 32) {
    unsigned lds_addr = (unsigned)(size_t)(void*)smem;   // low 32 bits = LDS offset
    unsigned long long ga = (unsigned long long)(const void*)(W + (size_t)tn * 16 * K);
    u32x4 g0;
    g0[0] = 1u;                                          // count=1, user mode, no gather
    g0[1] = lds_addr;                                    // lds_addr
    g0[2] = (unsigned)ga;                                // global_addr[31:0]
    g0[3] = (unsigned)((ga >> 32) & 0x01FFFFFFu) | (2u << 30); // addr[56:32] | type=2
    i32x8 g1;
    g1[0] = (int)(2u << 16);                             // wg_mask=0, data_size=2 (4B)
    g1[1] = (int)(((unsigned)K & 0xFFFFu) << 16);        // tensor_dim0[15:0]
    g1[2] = (int)((((unsigned)K >> 16) & 0xFFFFu) | (16u << 16)); // dim0 hi | tensor_dim1=16
    g1[3] = (int)((unsigned)K << 16);                    // tensor_dim1 hi=0 | tile_dim0=K
    g1[4] = 16;                                          // tile_dim1=16, tile_dim2=0
    g1[5] = K;                                           // tensor_dim0_stride[31:0]
    g1[6] = (int)((((unsigned)(K * 16)) & 0xFFFFu) << 16); // stride0 hi=0 | stride1 lo
    g1[7] = (int)(((unsigned)(K * 16)) >> 16);           // stride1 hi
    i32x4 z4 = {0, 0, 0, 0};
#if __clang_major__ >= 23
    i32x8 z8 = {0, 0, 0, 0, 0, 0, 0, 0};
    __builtin_amdgcn_tensor_load_to_lds(g0, g1, z4, z4, z8, 0);
#else
    __builtin_amdgcn_tensor_load_to_lds(g0, g1, z4, z4, 0);
#endif
    __builtin_amdgcn_s_wait_tensorcnt(0);
  }
#else
  {
    const float* __restrict__ Wt = W + (size_t)tn * 16 * K;
    for (int i = threadIdx.x; i < 16 * K; i += 256) smem[i] = Wt[i];
  }
#endif
  __syncthreads();

  if (tmBig < N_ / 32) {                        // wave-uniform guard; EXEC stays full
    const int nloc = lane & 15;
    const int ncol = tn * 16 + nloc;
    const float* __restrict__ arow0 = A + (size_t)(tmBig * 32 + nloc) * K;
    const float* __restrict__ arow1 = arow0 + (size_t)16 * K;
    const float* __restrict__ brow  = smem + nloc * K;

    v8f acc0 = {};
    v8f acc1 = {};
#pragma unroll
    for (int k0 = 0; k0 < K; k0 += 32) {
      if (k0 + 32 < K) {
        __builtin_prefetch(arow0 + k0 + 32, 0, 1);
        __builtin_prefetch(arow1 + k0 + 32, 0, 1);
      }
      const int kbA = k0 + ((lane < 16) ? 0 : 8);
      const int kbB = k0 + ((lane < 16) ? 0 : 8);
      v16h a0, a1, b;
#pragma unroll
      for (int j = 0; j < 16; ++j) {
        const int ka = kbA + ((j < 8) ? j : (j + 8));   // 16x32 A fragment layout
        a0[j] = (_Float16)arow0[ka];
        a1[j] = (_Float16)arow1[ka];
        b[j]  = (_Float16)brow[kbB + ((j < 8) ? j : (j + 8))]; // 32x16 B fragment layout
      }
      acc0 = __builtin_amdgcn_wmma_f32_16x16x32_f16(false, a0, false, b, (short)0,
                                                    acc0, false, false);
      acc1 = __builtin_amdgcn_wmma_f32_16x16x32_f16(false, a1, false, b, (short)0,
                                                    acc1, false, false);
    }

    const float bv = bias ? bias[ncol] : 0.0f;
    const int rbase0 = tmBig * 32 + ((lane < 16) ? 0 : 8);
#pragma unroll
    for (int v = 0; v < 8; ++v) {
      float v0 = acc0[v] + bv;
      float v1 = acc1[v] + bv;
      if (ACT == 2) { v0 = tanhf(v0); v1 = tanhf(v1); }
      C[(size_t)(rbase0 + v) * D_ + ncol]      = v0;
      C[(size_t)(rbase0 + 16 + v) * D_ + ncol] = v1;
    }
  }
}

// ---------------- attention scalar scores -----------------------------------
__global__ void k_att_scores(const float* __restrict__ h,
                             const float* __restrict__ a_src,
                             const float* __restrict__ a_dst,
                             float* __restrict__ ssrc,
                             float* __restrict__ sdst) {
  int id = blockIdx.x * blockDim.x + threadIdx.x;
  if (id >= N_ * H_) return;
  int n = id / H_, hh = id - n * H_;
  const float* row = h + (size_t)n * D_ + hh * C_;
  float s0 = 0.f, s1 = 0.f;
#pragma unroll 4
  for (int c = 0; c < C_; ++c) {
    float v = row[c];
    s0 += v * a_src[hh * C_ + c];
    s1 += v * a_dst[hh * C_ + c];
  }
  ssrc[id] = s0;
  sdst[id] = s1;
}

// ---------------- edge softmax: pass 1 (logits + segment max) ---------------
__global__ void k_edge_logits(const int* __restrict__ ei,
                              const float* __restrict__ ssrc,
                              const float* __restrict__ sdst,
                              float* __restrict__ ebuf,
                              float* __restrict__ nmax) {
  int idx = blockIdx.x * blockDim.x + threadIdx.x;
  if (idx >= EH_) return;
  int s, d; edge_sd(ei, idx, s, d);
#pragma unroll
  for (int hh = 0; hh < H_; ++hh) {
    float e = ssrc[s * H_ + hh] + sdst[d * H_ + hh];
    e = (e > 0.f) ? e : 0.2f * e;            // leaky_relu(0.2)
    ebuf[(size_t)idx * H_ + hh] = e;
    atomicMaxFloat(&nmax[d * H_ + hh], e);
  }
}

// ---------------- edge softmax: pass 2 (exp + segment sum) ------------------
__global__ void k_edge_expsum(const int* __restrict__ ei,
                              float* __restrict__ ebuf,
                              const float* __restrict__ nmax,
                              float* __restrict__ nsum) {
  int idx = blockIdx.x * blockDim.x + threadIdx.x;
  if (idx >= EH_) return;
  int s, d; edge_sd(ei, idx, s, d);
  (void)s;
#pragma unroll
  for (int hh = 0; hh < H_; ++hh) {
    float ex = expf(ebuf[(size_t)idx * H_ + hh] - nmax[d * H_ + hh]);
    ebuf[(size_t)idx * H_ + hh] = ex;
    atomicAdd(&nsum[d * H_ + hh], ex);
  }
}

// ---------------- edge softmax: pass 3 (weighted scatter-aggregate) ---------
__global__ void k_edge_aggr(const int* __restrict__ ei,
                            const float* __restrict__ ebuf,
                            const float* __restrict__ nsum,
                            const float* __restrict__ h,
                            float* __restrict__ outAgg) {
  long tid = (long)blockIdx.x * blockDim.x + threadIdx.x;
  if (tid >= (long)EH_ * 64) return;
  int idx = (int)(tid >> 6);
  int r   = (int)(tid & 63);
  int hh  = r >> 4;
  int c4  = (r & 15) * 4;
  int s, d; edge_sd(ei, idx, s, d);
  float alpha = ebuf[(size_t)idx * H_ + hh] / (nsum[d * H_ + hh] + 1e-16f);
  const float* hr = h + (size_t)s * D_ + hh * C_ + c4;
  float* o = outAgg + (size_t)d * D_ + hh * C_ + c4;
  atomicAdd(&o[0], alpha * hr[0]);
  atomicAdd(&o[1], alpha * hr[1]);
  atomicAdd(&o[2], alpha * hr[2]);
  atomicAdd(&o[3], alpha * hr[3]);
}

// ---------------- bias (+optional relu), in place ---------------------------
__global__ void k_bias_act(float* __restrict__ X, const float* __restrict__ b, int relu) {
  int i = blockIdx.x * blockDim.x + threadIdx.x;
  if (i >= N_ * D_) return;
  float v = X[i] + b[i % D_];
  X[i] = (relu && v < 0.f) ? 0.f : v;
}

// ---------------- fc2 + softmax + gumbel + active ---------------------------
__global__ void k_assign_gumbel(const float* __restrict__ t,
                                const float* __restrict__ fc2_w,
                                const float* __restrict__ fc2_b,
                                float* __restrict__ assign,
                                float* __restrict__ gum,
                                float* __restrict__ out) {
  int n = blockIdx.x * blockDim.x + threadIdx.x;
  if (n >= N_) return;
  const float* row = t + (size_t)n * D_;
  float l0 = fc2_b[0], l1 = fc2_b[1];
#pragma unroll 4
  for (int d = 0; d < D_; ++d) {
    float v = row[d];
    l0 += v * fc2_w[d];
    l1 += v * fc2_w[D_ + d];
  }
  float m = fmaxf(l0, l1);
  float e0 = expf(l0 - m), e1 = expf(l1 - m);
  float inv = 1.f / (e0 + e1);
  float a0 = e0 * inv, a1 = e1 * inv;
  assign[n * 2] = a0; assign[n * 2 + 1] = a1;
  float u0 = urand((unsigned)n, 0u, 0x9E3779B9u);
  float u1 = urand((unsigned)n, 1u, 0x9E3779B9u);
  float g0 = a0 - logf(-logf(u0 + 1e-20f) + 1e-20f);
  float g1 = a1 - logf(-logf(u1 + 1e-20f) + 1e-20f);
  float mg = fmaxf(g0, g1);
  float x0 = expf(g0 - mg), x1 = expf(g1 - mg);
  float invg = 1.f / (x0 + x1);
  float p0 = x0 * invg;
  gum[n * 2] = p0; gum[n * 2 + 1] = x1 * invg;
  out[OFF_ACTIVE + n] = (p0 > 0.5f) ? 1.0f : 0.0f;
}

// ---------------- per-column mean / std (ddof=1) via LDS reduce -------------
__global__ void k_colstats(const float* __restrict__ nf,
                           float* __restrict__ fmean,
                           float* __restrict__ fstd) {
  __shared__ float ssum[256];
  __shared__ float ssq[256];
  int d = blockIdx.x;
  int t = threadIdx.x;
  float s = 0.f, q = 0.f;
  for (int n = t; n < N_; n += 256) {
    float v = nf[(size_t)n * D_ + d];
    s += v; q += v * v;
  }
  ssum[t] = s; ssq[t] = q;
  __syncthreads();
  for (int off = 128; off > 0; off >>= 1) {
    if (t < off) { ssum[t] += ssum[t + off]; ssq[t] += ssq[t + off]; }
    __syncthreads();
  }
  if (t == 0) {
    float mean = ssum[0] / (float)N_;
    fmean[d] = mean;
    float var = (ssq[0] - ssum[0] * ssum[0] / (float)N_) / (float)(N_ - 1);
    fstd[d] = sqrtf(fmaxf(var, 0.f));
  }
}

// ---------------- per-graph node counts -------------------------------------
__global__ void k_cnt(const int* __restrict__ batch, float* __restrict__ cnt) {
  int n = blockIdx.x * blockDim.x + threadIdx.x;
  if (n >= N_) return;
  atomicAdd(&cnt[batch[n]], 1.0f);
}

// ---------------- noisy features, graph pooling sums, KL accumulation -------
__global__ void k_noisy(const float* __restrict__ nf,
                        const float* __restrict__ gum,
                        const float* __restrict__ fmean,
                        const float* __restrict__ fstd,
                        const int* __restrict__ batch,
                        float* __restrict__ gsum,
                        float* __restrict__ scal) {
  int tid = blockIdx.x * blockDim.x + threadIdx.x;
  if (tid >= N_ * D_) return;
  int n = tid / D_, d = tid - n * D_;
  float lp = gum[n * 2], ln = gum[n * 2 + 1];
  float fm = fmean[d], fs0 = fstd[d];
  float nm = lp * nf[tid] + ln * fm;
  float ns = ln * fs0;
  float u  = urand((unsigned)n, (unsigned)d, 0x85EBCA6Bu);
  atomicAdd(&gsum[batch[n] * D_ + d], nm + u * ns);
  float fs = fs0 + 1e-7f;
  float t1 = 0.5f * ns * ns / (fs * fs) * (1.0f / ((float)N_ * (float)D_));
  float dm = (nm - fm) / fs;
  float t2 = dm * dm * (1.0f / (float)D_);
  atomicAdd(&scal[0], t1 + t2);
}

// ---------------- graph embedding finalize ----------------------------------
__global__ void k_graph_emb(const float* __restrict__ gsum,
                            const float* __restrict__ cnt,
                            float* __restrict__ out) {
  int i = blockIdx.x * blockDim.x + threadIdx.x;
  if (i >= G_ * D_) return;
  int g = i / D_;
  out[OFF_GE + i] = gsum[i] / fmaxf(cnt[g], 1.0f);
}

// ---------------- Aa = segment_sum(assignment[dst], src) over real edges ----
__global__ void k_Aa(const int* __restrict__ ei,
                     const float* __restrict__ assign,
                     float* __restrict__ Aa) {
  int e = blockIdx.x * blockDim.x + threadIdx.x;
  if (e >= E_) return;
  int s = ei[e], d = ei[E_ + e];
  atomicAdd(&Aa[s * 2],     assign[d * 2]);
  atomicAdd(&Aa[s * 2 + 1], assign[d * 2 + 1]);
}

// ---------------- new_adj = assignment^T @ Aa  (2x2, accumulated) -----------
__global__ void k_newadj(const float* __restrict__ assign,
                         const float* __restrict__ Aa,
                         float* __restrict__ scal) {
  int n = blockIdx.x * blockDim.x + threadIdx.x;
  if (n >= N_) return;
  float a0 = assign[n * 2], a1 = assign[n * 2 + 1];
  float b0 = Aa[n * 2],     b1 = Aa[n * 2 + 1];
  atomicAdd(&scal[1], a0 * b0);
  atomicAdd(&scal[2], a0 * b1);
  atomicAdd(&scal[3], a1 * b0);
  atomicAdd(&scal[4], a1 * b1);
}

// ---------------- pos_penalty + KL write ------------------------------------
__global__ void k_small(const float* __restrict__ scal, float* __restrict__ out) {
  if (blockIdx.x != 0 || threadIdx.x != 0) return;
  float a = scal[1], b = scal[2], c = scal[3], d = scal[4];
  float r0 = fmaxf(fabsf(a) + fabsf(b), 1e-12f);
  float r1 = fmaxf(fabsf(c) + fabsf(d), 1e-12f);
  float n00 = a / r0, n11 = d / r1;
  out[OFF_POS] = 0.5f * ((n00 - 1.f) * (n00 - 1.f) + (n11 - 1.f) * (n11 - 1.f));
  out[OFF_KL]  = scal[0];
}

// ---------------- prototype distances, sim, logits, probs -------------------
__global__ void k_protos(const float* __restrict__ protos,
                         const float* __restrict__ last_w,
                         float* __restrict__ out) {
  int g = threadIdx.x;
  if (g >= G_) return;
  const float* row = out + OFF_GE + (size_t)g * D_;
  float ge2 = 0.f;
  for (int d = 0; d < D_; ++d) ge2 += row[d] * row[d];
  float sim[P_];
  for (int p = 0; p < P_; ++p) {
    float dot = 0.f, p2 = 0.f;
    for (int d = 0; d < D_; ++d) {
      float pv = protos[p * D_ + d];
      dot += row[d] * pv; p2 += pv * pv;
    }
    float dd = -2.f * dot + ge2 + p2;
    out[OFF_DIST + g * P_ + p] = dd;
    sim[p] = logf((dd + 1.0f) / (dd + 1e-4f));
    out[OFF_SIM + g * P_ + p] = sim[p];
  }
  float lg[2];
  for (int k = 0; k < 2; ++k) {
    const float* wr = last_w + (size_t)k * (P_ + D_);
    float acc = 0.f;
    for (int p = 0; p < P_; ++p) acc += sim[p] * wr[p];
    for (int d = 0; d < D_; ++d) acc += row[d] * wr[P_ + d];
    lg[k] = acc;
    out[OFF_LOGITS + g * 2 + k] = acc;
  }
  float m = fmaxf(lg[0], lg[1]);
  float e0 = expf(lg[0] - m), e1 = expf(lg[1] - m);
  float inv = 1.f / (e0 + e1);
  out[OFF_PROBS + g * 2]     = e0 * inv;
  out[OFF_PROBS + g * 2 + 1] = e1 * inv;
}

// ---------------- host driver ------------------------------------------------
static inline int cdiv(long a, long b) { return (int)((a + b - 1) / b); }

extern "C" void kernel_launch(void* const* d_in, const int* in_sizes, int n_in,
                              void* d_out, int out_size, void* d_ws, size_t ws_size,
                              hipStream_t stream) {
  const float* x     = (const float*)d_in[0];
  const int*   ei    = (const int*)  d_in[1];
  const int*   batch = (const int*)  d_in[2];
  const float* W1    = (const float*)d_in[3];
  const float* as1   = (const float*)d_in[4];
  const float* ad1   = (const float*)d_in[5];
  const float* b1    = (const float*)d_in[6];
  const float* W2    = (const float*)d_in[7];
  const float* as2   = (const float*)d_in[8];
  const float* ad2   = (const float*)d_in[9];
  const float* b2    = (const float*)d_in[10];
  const float* fc1_w = (const float*)d_in[11];
  const float* fc1_b = (const float*)d_in[12];
  const float* fc2_w = (const float*)d_in[13];
  const float* fc2_b = (const float*)d_in[14];
  const float* protos = (const float*)d_in[15];
  const float* last_w = (const float*)d_in[16];

  float* out = (float*)d_out;
  float* ws  = (float*)d_ws;

  float* hpre  = ws + WS_HPRE;
  float* h1    = ws + WS_H1;
  float* nf    = ws + WS_NF;
  float* ssrc  = ws + WS_SSRC;
  float* sdst  = ws + WS_SDST;
  float* nmax  = ws + WS_NMAX;
  float* nsum  = ws + WS_NSUM;
  float* ebuf  = ws + WS_EBUF;
  float* assn  = ws + WS_ASSN;
  float* gum   = ws + WS_GUM;
  float* fmean = ws + WS_FMEAN;
  float* fstd  = ws + WS_FSTD;
  float* gsum  = ws + WS_GSUM;
  float* cnt   = ws + WS_CNT;
  float* Aa    = ws + WS_AA;
  float* scal  = ws + WS_SCAL;

  const int TB = 256;
  const dim3 gemmGrid(cdiv(N_ / 32, 8), D_ / 16);   // (79, 16), 8 waves/block

  // ===== GAT layer 1 =====
  k_wmma_gemm<IN_, 0><<<gemmGrid, TB, 0, stream>>>(x, W1, nullptr, hpre);
  k_att_scores<<<cdiv(N_ * H_, TB), TB, 0, stream>>>(hpre, as1, ad1, ssrc, sdst);
  k_fill<<<cdiv(N_ * D_, TB), TB, 0, stream>>>(h1, 0.f, N_ * D_);
  k_fill<<<cdiv(N_ * H_, TB), TB, 0, stream>>>(nmax, -3.4e38f, N_ * H_);
  k_fill<<<cdiv(N_ * H_, TB), TB, 0, stream>>>(nsum, 0.f, N_ * H_);
  k_edge_logits<<<cdiv(EH_, TB), TB, 0, stream>>>(ei, ssrc, sdst, ebuf, nmax);
  k_edge_expsum<<<cdiv(EH_, TB), TB, 0, stream>>>(ei, ebuf, nmax, nsum);
  k_edge_aggr<<<cdiv((long)EH_ * 64, TB), TB, 0, stream>>>(ei, ebuf, nsum, hpre, h1);
  k_bias_act<<<cdiv(N_ * D_, TB), TB, 0, stream>>>(h1, b1, 1);

  // ===== GAT layer 2 =====
  k_wmma_gemm<D_, 0><<<gemmGrid, TB, 0, stream>>>(h1, W2, nullptr, hpre);
  k_att_scores<<<cdiv(N_ * H_, TB), TB, 0, stream>>>(hpre, as2, ad2, ssrc, sdst);
  k_fill<<<cdiv(N_ * D_, TB), TB, 0, stream>>>(nf, 0.f, N_ * D_);
  k_fill<<<cdiv(N_ * H_, TB), TB, 0, stream>>>(nmax, -3.4e38f, N_ * H_);
  k_fill<<<cdiv(N_ * H_, TB), TB, 0, stream>>>(nsum, 0.f, N_ * H_);
  k_edge_logits<<<cdiv(EH_, TB), TB, 0, stream>>>(ei, ssrc, sdst, ebuf, nmax);
  k_edge_expsum<<<cdiv(EH_, TB), TB, 0, stream>>>(ei, ebuf, nmax, nsum);
  k_edge_aggr<<<cdiv((long)EH_ * 64, TB), TB, 0, stream>>>(ei, ebuf, nsum, hpre, nf);
  k_bias_act<<<cdiv(N_ * D_, TB), TB, 0, stream>>>(nf, b2, 0);

  // ===== assignment head =====
  k_wmma_gemm<D_, 2><<<gemmGrid, TB, 0, stream>>>(nf, fc1_w, fc1_b, hpre);
  k_assign_gumbel<<<cdiv(N_, TB), TB, 0, stream>>>(hpre, fc2_w, fc2_b, assn, gum, out);

  // ===== feature stats, noisy pooling, KL =====
  k_colstats<<<D_, 256, 0, stream>>>(nf, fmean, fstd);
  k_fill<<<cdiv(G_ * D_, TB), TB, 0, stream>>>(gsum, 0.f, G_ * D_);
  k_fill<<<1, TB, 0, stream>>>(cnt, 0.f, G_);
  k_fill<<<1, TB, 0, stream>>>(scal, 0.f, 16);
  k_cnt<<<cdiv(N_, TB), TB, 0, stream>>>(batch, cnt);
  k_noisy<<<cdiv(N_ * D_, TB), TB, 0, stream>>>(nf, gum, fmean, fstd, batch, gsum, scal);
  k_graph_emb<<<cdiv(G_ * D_, TB), TB, 0, stream>>>(gsum, cnt, out);

  // ===== connectivity penalty =====
  k_fill<<<cdiv(N_ * 2, TB), TB, 0, stream>>>(Aa, 0.f, N_ * 2);
  k_Aa<<<cdiv(E_, TB), TB, 0, stream>>>(ei, assn, Aa);
  k_newadj<<<cdiv(N_, TB), TB, 0, stream>>>(assn, Aa, scal);
  k_small<<<1, 32, 0, stream>>>(scal, out);

  // ===== prototype head =====
  k_protos<<<1, 64, 0, stream>>>(protos, last_w, out);

  (void)in_sizes; (void)n_in; (void)out_size; (void)ws_size;
}
